// ContrastiveTwins_37589553774707
// MI455X (gfx1250) — matrix operand order repaired and checked by reading, not compiled, not run on hardware
//
#include <hip/hip_runtime.h>
#include <hip/hip_bf16.h>

// ---------------------------------------------------------------------------
// BERT-base 12-layer encoder forward for MI455X (gfx1250, wave32, WMMA).
// All GEMMs run as bf16 NT GEMMs through V_WMMA_F32_16X16X32_BF16.
// Each wave computes a 32x64 output tile (2x4 WMMA subtiles).  K loop is
// unrolled by 2 with ping-ponged fragment register sets (no copy movs):
// steady state = 24 global_load_b128 + 16 WMMAs per iteration.
// Requires Kd % 64 == 0 (true for all GEMMs here: 768/64/512/3072).
// ---------------------------------------------------------------------------

typedef __bf16 bf16_t;
typedef __attribute__((ext_vector_type(16))) __bf16 v16bf;
typedef __attribute__((ext_vector_type(8)))  __bf16 v8bf;
typedef __attribute__((ext_vector_type(8)))  float  v8f;

constexpr int LAYERS = 12;
constexpr int H      = 768;
constexpr int NHEADS = 12;
constexpr int DHEAD  = 64;
constexpr int BATCH  = 8;
constexpr int SEQ    = 512;
constexpr int INTER  = 3072;
constexpr int MROWS  = BATCH * SEQ;     // 4096
constexpr float ATT_SCALE = 0.125f;     // 1/sqrt(64)
constexpr float LNEPS     = 1e-12f;

// ---------------------------------------------------------------------------
// Batched NT GEMM:  C[M,N] (f32) = A[M,K](bf16, row-major, lda) x Bt[N,K](bf16,
// row-major, ldb == B column-major).
//
// Per-lane fragment layouts used here (contiguous K order, not the ISA table order; A and B match):
//   A: row M = lane%16.  lanes 0-15 hold K k0+0..15, lanes 16-31 hold K k0+16..31
//      -> one contiguous 32B load at (k0 + half*16).
//   B: col N = lane%16.  lanes 0-15 hold K k0+0..15, lanes 16-31 hold K k0+16..31
//      -> one contiguous 32B load at (k0 + half*16).
//   C: col N = lane%16, VGPR r -> row m0 + r + 8*(lane/16).
//
// Wave tile: 32 rows (2 subtiles) x 64 cols (4 subtiles), 8 accumulators.
// Block: 4 waves stacked in M -> block tile 128 x 64.
// ---------------------------------------------------------------------------
__global__ __launch_bounds__(128)
void gemm_nt_bf16_kernel(const bf16_t* __restrict__ A,
                         const bf16_t* __restrict__ Bt,
                         float* __restrict__ C,
                         int lda, int ldb, int ldc, int Kd, int nhDiv,
                         long long sA1, long long sA2,
                         long long sB1, long long sB2,
                         long long sC1, long long sC2)
{
    const int z  = blockIdx.z;
    const int zq = z / nhDiv;
    const int zr = z % nhDiv;
    const long long offA = (long long)zq * sA1 + (long long)zr * sA2;
    const long long offB = (long long)zq * sB1 + (long long)zr * sB2;
    const long long offC = (long long)zq * sC1 + (long long)zr * sC2;

    const int lane = threadIdx.x & 31;
    const int wave = threadIdx.x >> 5;
    const int hl   = lane >> 4;        // 0: lanes 0-15, 1: lanes 16-31
    const int lm   = lane & 15;

    const int m0 = blockIdx.y * 128 + wave * 32;
    const int n0 = blockIdx.x * 64;

    const bf16_t* Ar0 = A  + offA + (long long)(m0 + lm) * lda;        // rows m0..m0+15
    const bf16_t* Ar1 = Ar0 + (long long)16 * lda;                     // rows m0+16..m0+31
    const bf16_t* Bc  = Bt + offB + (long long)(n0 + lm) * ldb;        // cols n0..n0+15 (+16*ni)

    v8f acc[8];
#pragma unroll
    for (int i = 0; i < 8; ++i)
        acc[i] = (v8f){0.f, 0.f, 0.f, 0.f, 0.f, 0.f, 0.f, 0.f};

    v16bf a0[2], b0[4], a1[2], b1[4];   // ping-pong fragment sets

    auto loadFrags = [&](v16bf* da, v16bf* db, int k0) {
#pragma unroll
        for (int mi = 0; mi < 2; ++mi) {
            const bf16_t* p = (mi ? Ar1 : Ar0) + k0 + hl * 16;
            da[mi] = *(const v16bf*)p;
        }
#pragma unroll
        for (int ni = 0; ni < 4; ++ni)
            db[ni] = *(const v16bf*)(Bc + (long long)ni * 16 * ldb + k0 + hl * 16);
    };

    auto wmma8 = [&](const v16bf* a, const v16bf* b) {
#pragma unroll
        for (int mi = 0; mi < 2; ++mi)
#pragma unroll
            for (int ni = 0; ni < 4; ++ni)
                acc[mi * 4 + ni] = __builtin_amdgcn_wmma_f32_16x16x32_bf16(
                    false, a[mi], false, b[ni], (short)0, acc[mi * 4 + ni],
                    false, false);
    };

    loadFrags(a0, b0, 0);

    // Kd % 64 == 0 assumed; 2 K-slices (2x8 WMMAs) per iteration, ping-pong.
    for (int k0 = 0; k0 < Kd; k0 += 64) {
        const bool more = (k0 + 64) < Kd;
        loadFrags(a1, b1, k0 + 32);                 // set 1 <- k0+32
        if (more) {
            __builtin_prefetch(Ar0 + k0 + 96, 0, 3); // global_prefetch_b8
            __builtin_prefetch(Bc  + k0 + 96, 0, 3);
        }
        wmma8(a0, b0);                              // compute k0
        if (more)
            loadFrags(a0, b0, k0 + 64);             // set 0 <- k0+64
        wmma8(a1, b1);                              // compute k0+32
    }

#pragma unroll
    for (int mi = 0; mi < 2; ++mi) {
#pragma unroll
        for (int ni = 0; ni < 4; ++ni) {
            float* Crow = C + offC
                        + (long long)(m0 + mi * 16 + hl * 8) * ldc
                        + n0 + ni * 16 + lm;
            v8f v = acc[mi * 4 + ni];
#pragma unroll
            for (int r = 0; r < 8; ++r)
                Crow[(long long)r * ldc] = v[r];
        }
    }
}

// ---------------------------------------------------------------------------
// Weight transpose + f32->bf16 cast: in[l][K,N] f32  ->  out[l][N,K] bf16.
// ---------------------------------------------------------------------------
__global__ __launch_bounds__(256)
void wcast_kernel(const float* __restrict__ in, bf16_t* __restrict__ out,
                  int K, int N, long long inL, long long outL)
{
    long long idx = (long long)blockIdx.x * blockDim.x + threadIdx.x;
    long long tot = (long long)K * N;
    if (idx >= tot) return;
    int l = blockIdx.z;
    int k = (int)(idx / N);
    int n = (int)(idx % N);
    out[(long long)l * outL + (long long)n * K + k] =
        (bf16_t)in[(long long)l * inL + idx];
}

__global__ __launch_bounds__(256)
void prep_hidden_kernel(const float* __restrict__ in, float* __restrict__ xf,
                        bf16_t* __restrict__ xb, long long tot)
{
    long long i = (long long)blockIdx.x * blockDim.x + threadIdx.x;
    if (i >= tot) return;
    float v = in[i];
    xf[i] = v;
    xb[i] = (bf16_t)v;
}

__global__ __launch_bounds__(256)
void bias_cast_kernel(const float* __restrict__ g, const float* __restrict__ bias,
                      bf16_t* __restrict__ out, int ncol, long long tot)
{
    long long i = (long long)blockIdx.x * blockDim.x + threadIdx.x;
    if (i >= tot) return;
    out[i] = (bf16_t)(g[i] + bias[i % ncol]);
}

// v[B,S,H] + bias  ->  vT[B,NH,DH,S]  (bf16), so P@V is an NT GEMM.
__global__ __launch_bounds__(256)
void vT_kernel(const float* __restrict__ vg, const float* __restrict__ bias,
               bf16_t* __restrict__ vT)
{
    long long i = (long long)blockIdx.x * blockDim.x + threadIdx.x;
    const long long tot = (long long)BATCH * NHEADS * DHEAD * SEQ;
    if (i >= tot) return;
    int s = (int)(i % SEQ);
    int d = (int)((i / SEQ) % DHEAD);
    int h = (int)((i / ((long long)SEQ * DHEAD)) % NHEADS);
    int b = (int)(i / ((long long)SEQ * DHEAD * NHEADS));
    vT[i] = (bf16_t)(vg[((long long)(b * SEQ + s)) * H + h * DHEAD + d]
                     + bias[h * DHEAD + d]);
}

// ctx[B,NH,S,DH] f32  ->  cb[B,S,H] bf16 (head merge for the O projection).
__global__ __launch_bounds__(256)
void ctx_merge_kernel(const float* __restrict__ ctxg, bf16_t* __restrict__ cb)
{
    long long i = (long long)blockIdx.x * blockDim.x + threadIdx.x;
    const long long tot = (long long)BATCH * SEQ * H;
    if (i >= tot) return;
    int d = (int)(i % DHEAD);
    int h = (int)((i / DHEAD) % NHEADS);
    int s = (int)((i / ((long long)DHEAD * NHEADS)) % SEQ);
    int b = (int)(i / ((long long)DHEAD * NHEADS * SEQ));
    cb[i] = (bf16_t)ctxg[(((long long)(b * NHEADS + h)) * SEQ + s) * DHEAD + d];
}

// Row softmax over scores[B*NH*S, S] with scale + additive mask; bf16 output.
__global__ __launch_bounds__(128)
void softmax_kernel(const float* __restrict__ scores,
                    const float* __restrict__ mask,
                    bf16_t* __restrict__ probs)
{
    long long row = blockIdx.x;                       // [0, B*NH*S)
    int b = (int)(row / ((long long)NHEADS * SEQ));
    const float* srow = scores + row * SEQ;
    const float* mrow = mask + (long long)b * SEQ;
    bf16_t* prow = probs + row * SEQ;
    int t = threadIdx.x;

    float vals[4];
    float mx = -3.0e38f;
#pragma unroll
    for (int i = 0; i < 4; ++i) {
        int j = t + i * 128;
        float v = srow[j] * ATT_SCALE + (1.0f - mrow[j]) * -10000.0f;
        vals[i] = v;
        mx = fmaxf(mx, v);
    }
    __shared__ float red[128];
    red[t] = mx; __syncthreads();
    for (int w = 64; w > 0; w >>= 1) {
        if (t < w) red[t] = fmaxf(red[t], red[t + w]);
        __syncthreads();
    }
    mx = red[0]; __syncthreads();

    float sum = 0.f;
#pragma unroll
    for (int i = 0; i < 4; ++i) { vals[i] = __expf(vals[i] - mx); sum += vals[i]; }
    red[t] = sum; __syncthreads();
    for (int w = 64; w > 0; w >>= 1) {
        if (t < w) red[t] += red[t + w];
        __syncthreads();
    }
    float inv = 1.0f / red[0];
#pragma unroll
    for (int i = 0; i < 4; ++i)
        prow[t + i * 128] = (bf16_t)(vals[i] * inv);
}

// out = LayerNorm(g + bias + resid) * gamma + beta; writes f32 + bf16 copies.
__global__ __launch_bounds__(256)
void bias_res_ln_kernel(const float* __restrict__ g, const float* __restrict__ bias,
                        const float* __restrict__ resid,
                        const float* __restrict__ gamma, const float* __restrict__ beta,
                        float* __restrict__ outF, bf16_t* __restrict__ outB)
{
    long long row = blockIdx.x;                       // [0, B*S)
    const float* grow = g + row * H;
    const float* rrow = resid + row * H;
    int t = threadIdx.x;

    float x[3];
    float s = 0.f, ss = 0.f;
#pragma unroll
    for (int i = 0; i < 3; ++i) {
        int j = t + i * 256;
        float v = grow[j] + bias[j] + rrow[j];
        x[i] = v; s += v; ss += v * v;
    }
    __shared__ float r1[256], r2[256];
    r1[t] = s; r2[t] = ss; __syncthreads();
    for (int w = 128; w > 0; w >>= 1) {
        if (t < w) { r1[t] += r1[t + w]; r2[t] += r2[t + w]; }
        __syncthreads();
    }
    float mean = r1[0] * (1.0f / H);
    float var  = r2[0] * (1.0f / H) - mean * mean;
    float inv  = rsqrtf(var + LNEPS);
#pragma unroll
    for (int i = 0; i < 3; ++i) {
        int j = t + i * 256;
        float y = (x[i] - mean) * inv * gamma[j] + beta[j];
        outF[row * H + j] = y;
        outB[row * H + j] = (bf16_t)y;
    }
}

// Exact-erf GELU with bias, f32 in -> bf16 out.
__global__ __launch_bounds__(256)
void gelu_kernel(const float* __restrict__ g, const float* __restrict__ bias,
                 bf16_t* __restrict__ out, int ncol, long long tot)
{
    long long i = (long long)blockIdx.x * blockDim.x + threadIdx.x;
    if (i >= tot) return;
    float v = g[i] + bias[i % ncol];
    out[i] = (bf16_t)(0.5f * v * (1.0f + erff(v * 0.70710678118f)));
}

// ---------------------------------------------------------------------------
extern "C" void kernel_launch(void* const* d_in, const int* in_sizes, int n_in,
                              void* d_out, int out_size, void* d_ws, size_t ws_size,
                              hipStream_t stream)
{
    (void)in_sizes; (void)n_in; (void)out_size; (void)ws_size;

    const float* hid   = (const float*)d_in[0];
    const float* amask = (const float*)d_in[1];
    const float* Wq  = (const float*)d_in[2];  const float* bq  = (const float*)d_in[3];
    const float* Wk  = (const float*)d_in[4];  const float* bk  = (const float*)d_in[5];
    const float* Wv  = (const float*)d_in[6];  const float* bv  = (const float*)d_in[7];
    const float* Wo  = (const float*)d_in[8];  const float* bo  = (const float*)d_in[9];
    const float* g1w = (const float*)d_in[10]; const float* b1w = (const float*)d_in[11];
    const float* Wi  = (const float*)d_in[12]; const float* bi  = (const float*)d_in[13];
    const float* Wf  = (const float*)d_in[14]; const float* bfv = (const float*)d_in[15];
    const float* g2w = (const float*)d_in[16]; const float* b2w = (const float*)d_in[17];

    char* ws = (char*)d_ws;
    size_t off = 0;
    auto alloc = [&](size_t bytes) -> void* {
        void* p = ws + off;
        off = (off + bytes + 255) & ~(size_t)255;
        return p;
    };
    const size_t HH = (size_t)H * H;
    const size_t HI = (size_t)H * INTER;
    const long long tMH = (long long)MROWS * H;
    const long long tMI = (long long)MROWS * INTER;
    const long long tSC = (long long)BATCH * NHEADS * SEQ * SEQ;

    bf16_t* WqT = (bf16_t*)alloc(2 * LAYERS * HH);
    bf16_t* WkT = (bf16_t*)alloc(2 * LAYERS * HH);
    bf16_t* WvT = (bf16_t*)alloc(2 * LAYERS * HH);
    bf16_t* WoT = (bf16_t*)alloc(2 * LAYERS * HH);
    bf16_t* WiT = (bf16_t*)alloc(2 * LAYERS * HI);   // [I,H] per layer
    bf16_t* WfT = (bf16_t*)alloc(2 * LAYERS * HI);   // [H,I] per layer

    float*  xf    = (float*)alloc(4 * (size_t)tMH);
    bf16_t* xb    = (bf16_t*)alloc(2 * (size_t)tMH);
    float*  qg    = (float*)alloc(4 * (size_t)tMH);
    float*  kg    = (float*)alloc(4 * (size_t)tMH);
    float*  vg    = (float*)alloc(4 * (size_t)tMH);
    bf16_t* qb    = (bf16_t*)alloc(2 * (size_t)tMH);
    bf16_t* kb    = (bf16_t*)alloc(2 * (size_t)tMH);
    bf16_t* vTb   = (bf16_t*)alloc(2 * (size_t)tMH);
    float*  scor  = (float*)alloc(4 * (size_t)tSC);
    bf16_t* probs = (bf16_t*)alloc(2 * (size_t)tSC);
    float*  ctxg  = (float*)alloc(4 * (size_t)tMH);
    bf16_t* cb    = (bf16_t*)alloc(2 * (size_t)tMH);
    float*  go    = (float*)alloc(4 * (size_t)tMH);
    float*  attnF = (float*)alloc(4 * (size_t)tMH);
    bf16_t* attnB = (bf16_t*)alloc(2 * (size_t)tMH);
    float*  ffn1  = (float*)alloc(4 * (size_t)tMI);
    bf16_t* h1b   = (bf16_t*)alloc(2 * (size_t)tMI);
    float*  ffn2  = (float*)alloc(4 * (size_t)tMH);

    auto cdiv = [](long long a, long long b) { return (unsigned)((a + b - 1) / b); };

    // One-time (per launch) weight transpose+cast to bf16 NT layout.
    {
        dim3 blk(256);
        dim3 gHH(cdiv((long long)HH, 256), 1, LAYERS);
        wcast_kernel<<<gHH, blk, 0, stream>>>(Wq, WqT, H, H, (long long)HH, (long long)HH);
        wcast_kernel<<<gHH, blk, 0, stream>>>(Wk, WkT, H, H, (long long)HH, (long long)HH);
        wcast_kernel<<<gHH, blk, 0, stream>>>(Wv, WvT, H, H, (long long)HH, (long long)HH);
        wcast_kernel<<<gHH, blk, 0, stream>>>(Wo, WoT, H, H, (long long)HH, (long long)HH);
        dim3 gHI(cdiv((long long)HI, 256), 1, LAYERS);
        wcast_kernel<<<gHI, blk, 0, stream>>>(Wi, WiT, H, INTER, (long long)HI, (long long)HI);
        wcast_kernel<<<gHI, blk, 0, stream>>>(Wf, WfT, INTER, H, (long long)HI, (long long)HI);
    }
    prep_hidden_kernel<<<cdiv(tMH, 256), 256, 0, stream>>>(hid, xf, xb, tMH);

    // Block tile: 128(M) x 64(N); wave tile 32x64.
    auto gemm = [&](const bf16_t* A, const bf16_t* Bt, float* C,
                    int Mr, int Nc, int Kd, int lda, int ldb, int ldc,
                    int batches, int nhDiv,
                    long long sA1, long long sA2, long long sB1, long long sB2,
                    long long sC1, long long sC2) {
        dim3 grid((unsigned)(Nc / 64), (unsigned)(Mr / 128), (unsigned)batches);
        gemm_nt_bf16_kernel<<<grid, dim3(128), 0, stream>>>(
            A, Bt, C, lda, ldb, ldc, Kd, nhDiv, sA1, sA2, sB1, sB2, sC1, sC2);
    };

    const long long SH = (long long)SEQ * H;
    const long long SS = (long long)SEQ * SEQ;
    const long long DS = (long long)DHEAD * SEQ;
    const int BH = BATCH * NHEADS;

    for (int l = 0; l < LAYERS; ++l) {
        const bf16_t* wq = WqT + (size_t)l * HH;
        const bf16_t* wk = WkT + (size_t)l * HH;
        const bf16_t* wv = WvT + (size_t)l * HH;
        const bf16_t* wo = WoT + (size_t)l * HH;
        const bf16_t* wi = WiT + (size_t)l * HI;
        const bf16_t* wf = WfT + (size_t)l * HI;
        const float* bql = bq  + (size_t)l * H;
        const float* bkl = bk  + (size_t)l * H;
        const float* bvl = bv  + (size_t)l * H;
        const float* bol = bo  + (size_t)l * H;
        const float* g1l = g1w + (size_t)l * H;
        const float* b1l = b1w + (size_t)l * H;
        const float* bil = bi  + (size_t)l * INTER;
        const float* bfl = bfv + (size_t)l * H;
        const float* g2l = g2w + (size_t)l * H;
        const float* b2l = b2w + (size_t)l * H;

        // Q/K/V projections: [4096,768] = [4096,768] x [768,768]
        gemm(xb, wq, qg, MROWS, H, H, H, H, H, 1, 1, 0, 0, 0, 0, 0, 0);
        gemm(xb, wk, kg, MROWS, H, H, H, H, H, 1, 1, 0, 0, 0, 0, 0, 0);
        gemm(xb, wv, vg, MROWS, H, H, H, H, H, 1, 1, 0, 0, 0, 0, 0, 0);
        bias_cast_kernel<<<cdiv(tMH, 256), 256, 0, stream>>>(qg, bql, qb, H, tMH);
        bias_cast_kernel<<<cdiv(tMH, 256), 256, 0, stream>>>(kg, bkl, kb, H, tMH);
        vT_kernel<<<cdiv(tMH, 256), 256, 0, stream>>>(vg, bvl, vTb);

        // scores[b,h] = Q_head @ K_head^T : batched NT (A=q, Bt=k), 96 batches
        gemm(qb, kb, scor, SEQ, SEQ, DHEAD, H, H, SEQ, BH, NHEADS,
             SH, DHEAD, SH, DHEAD, (long long)NHEADS * SS, SS);
        softmax_kernel<<<(unsigned)((long long)BH * SEQ), 128, 0, stream>>>(scor, amask, probs);

        // ctx[b,h] = P @ V : A = probs [S,S], Bt = V^T [DH,S]
        gemm(probs, vTb, ctxg, SEQ, DHEAD, SEQ, SEQ, SEQ, DHEAD, BH, NHEADS,
             (long long)NHEADS * SS, SS, (long long)NHEADS * DS, DS,
             (long long)NHEADS * (long long)SEQ * DHEAD, (long long)SEQ * DHEAD);
        ctx_merge_kernel<<<cdiv(tMH, 256), 256, 0, stream>>>(ctxg, cb);

        // O projection + residual + LN1
        gemm(cb, wo, go, MROWS, H, H, H, H, H, 1, 1, 0, 0, 0, 0, 0, 0);
        bias_res_ln_kernel<<<(unsigned)MROWS, 256, 0, stream>>>(
            go, bol, xf, g1l, b1l, attnF, attnB);

        // FFN: [4096,3072] then [4096,768], residual + LN2
        gemm(attnB, wi, ffn1, MROWS, INTER, H, H, H, INTER, 1, 1, 0, 0, 0, 0, 0, 0);
        gelu_kernel<<<cdiv(tMI, 256), 256, 0, stream>>>(ffn1, bil, h1b, INTER, tMI);
        gemm(h1b, wf, ffn2, MROWS, H, INTER, INTER, INTER, H, 1, 1, 0, 0, 0, 0, 0, 0);

        float* outF = (l == LAYERS - 1) ? (float*)d_out : xf;
        bias_res_ln_kernel<<<(unsigned)MROWS, 256, 0, stream>>>(
            ffn2, bfl, attnF, g2l, b2l, outF, xb);
    }
}
